// MCModel_16956530884805
// MI455X (gfx1250) — matrix operand order repaired and checked by reading, not compiled, not run on hardware
//
#include <hip/hip_runtime.h>
#include <hip/hip_bf16.h>

typedef __bf16 bf16x16 __attribute__((ext_vector_type(16)));
typedef float  f32x8   __attribute__((ext_vector_type(8)));

union Frag {
  uint4   q[2];
  bf16x16 v;
};

#define B_  64
#define T_  256
#define M_  16
#define A_  32
#define H_  64

// workspace (ushort element offsets): bf16 copies of the weights
#define WS_WR 0
#define WS_WJ 131072
#define WS_WO (131072 + 32768)
#define W_TOTAL (131072 + 32768 + 2048)

__device__ __forceinline__ unsigned short f2bf(float f) {
  unsigned int u = __float_as_uint(f);
  u += 0x7FFFu + ((u >> 16) & 1u);   // round-to-nearest-even
  return (unsigned short)(u >> 16);
}

// single-instruction v_rcp_f32 (safe here: denominators are >= 1 or well-scaled)
__device__ __forceinline__ float fast_rcp(float x) {
  return __builtin_amdgcn_rcpf(x);
}

__global__ void wconv_kernel(const float* __restrict__ Wr, const float* __restrict__ Wj,
                             const float* __restrict__ Wo, unsigned short* __restrict__ wbf) {
  int i = blockIdx.x * 256 + threadIdx.x;
  if (i < 131072)                       wbf[WS_WR + i] = f2bf(Wr[i]);
  else if (i < 131072 + 32768)          wbf[WS_WJ + (i - 131072)] = f2bf(Wj[i - 131072]);
  else if (i < W_TOTAL)                 wbf[WS_WO + (i - 131072 - 32768)] = f2bf(Wo[i - 131072 - 32768]);
}

// Dynamic LDS layout (bytes):
//   Rbuf  f32 [16][4096]  @ 0        (262144)   first [16][1024] aliased as Jbuf
//   XA    bf16[16][32]    @ 262144   (1024)
//   XM    f32 [16][16]    @ 263168   (1024)
//   OgS   f32 [16][64]    @ 264192   (4096)
//   MinS  f32 [16][64]    @ 268288   (4096)
//   Smax  f32 [256]       @ 272384   (1024)
//   Srd   f32 [256]       @ 273408   (1024)
//   Cst   f32 [64]        @ 274432   (256)
//   Mout  f32 [64]        @ 274688   (256)
#define LDS_BYTES 274944

__launch_bounds__(256, 1)
__global__ void mclstm_kernel(const float* __restrict__ x_m, const float* __restrict__ x_a,
                              const float* __restrict__ bj, const float* __restrict__ br,
                              const float* __restrict__ bo, const float* __restrict__ Wfc,
                              const float* __restrict__ bfc,
                              const unsigned short* __restrict__ wbf,
                              float* __restrict__ out) {
  extern __shared__ char smem[];
  float*          Rbuf = (float*)smem;
  unsigned short* XA   = (unsigned short*)(smem + 262144);
  float*          XM   = (float*)(smem + 263168);
  float*          OgS  = (float*)(smem + 264192);
  float*          MinS = (float*)(smem + 268288);
  float*          Smax = (float*)(smem + 272384);
  float*          Srd  = (float*)(smem + 273408);
  float*          Cst  = (float*)(smem + 274432);
  float*          Mout = (float*)(smem + 274688);

  const int tid  = threadIdx.x;
  const int lane = tid & 31;
  const int wv   = tid >> 5;
  const int b    = blockIdx.x;

  if (tid < 64) { Cst[tid] = 0.f; Mout[tid] = 0.f; }
  __syncthreads();

  // WMMA lane geometry (16x16x32 bf16, wave32)
  const int row16 = lane & 15;
  const int akb   = (lane < 16) ? 0 : 8;   // A-matrix K base (two runs: akb, akb+16)
  const int bkb   = (lane < 16) ? 0 : 8;   // B-matrix K base (two runs: bkb, bkb+16)
  const int moff  = (lane < 16) ? 0 : 8;   // C/D row base

  for (int ci = 0; ci < 16; ++ci) {
    const int t0 = ci * 16;

    // ---- stage x_a (bf16) and x_m (f32) chunk into LDS ----
    for (int i = tid; i < 512; i += 256) {
      int tt = i >> 5, aa = i & 31;
      XA[i] = f2bf(x_a[((size_t)b * T_ + (t0 + tt)) * A_ + aa]);
    }
    {
      int tt = tid >> 4, mm = tid & 15;
      XM[tid] = x_m[((size_t)b * T_ + (t0 + tt)) * M_ + mm];
    }
    __syncthreads();

    // ---- A fragment: shared by every tile this chunk ----
    Frag afr;
    afr.q[0] = *(const uint4*)&XA[row16 * 32 + akb];
    afr.q[1] = *(const uint4*)&XA[row16 * 32 + akb + 16];

    // ---- Phase A: J logits (tiles 0..63) + out-gate (tiles 64..67) ----
    for (int tile = wv; tile < 68; tile += 8) {
      const bool isOg = (tile >= 64);
      const unsigned short* wb = isOg ? (wbf + WS_WO) : (wbf + WS_WJ);
      const float* bias        = isOg ? bo : bj;
      const int n0             = isOg ? (tile - 64) * 16 : tile * 16;
      const int colg = n0 + row16;

      Frag bfr;
      const unsigned short* wp = wb + colg * 32 + bkb;
      bfr.q[0] = *(const uint4*)wp;
      bfr.q[1] = *(const uint4*)(wp + 16);

      f32x8 acc = {};
      acc = __builtin_amdgcn_wmma_f32_16x16x32_bf16(false, afr.v, false, bfr.v,
                                                    (short)0, acc, false, false);
      const float bv = bias[colg];
      if (!isOg) {
        for (int r = 0; r < 8; ++r)
          Rbuf[(r + moff) * 1024 + colg] = acc[r] + bv;        // Jbuf alias
      } else {
        for (int r = 0; r < 8; ++r) {
          float v = acc[r] + bv;
          OgS[(r + moff) * 64 + colg] = fast_rcp(1.f + __expf(-v));
        }
      }
    }
    __syncthreads();

    // ---- J softmax stats per (t,m), bank-skewed walk ----
    {
      int tt = tid >> 4, mm = tid & 15;
      const float* base = Rbuf + tt * 1024 + mm * 64;
      int k0 = tid & 63;
      float mx = -1e30f;
      for (int i = 0; i < 64; ++i) { int k = (k0 + i) & 63; mx = fmaxf(mx, base[k]); }
      float den = 0.f;
      for (int i = 0; i < 64; ++i) { int k = (k0 + i) & 63; den += __expf(base[k] - mx); }
      Smax[tid] = mx;
      Srd[tid]  = fast_rcp(den);
    }
    __syncthreads();

    // ---- m_in[t][h] = sum_m x_m * softmax(J) ----
    for (int p = tid; p < 1024; p += 256) {
      int tt = p >> 6, hh = p & 63;
      float acc = 0.f;
      const float* jcol = Rbuf + tt * 1024 + hh;
      for (int mm = 0; mm < 16; ++mm) {
        int tm = tt * 16 + mm;
        acc += XM[tm] * __expf(jcol[mm * 64] - Smax[tm]) * Srd[tm];
      }
      MinS[tt * 64 + hh] = acc;
    }
    __syncthreads();

    // ---- Phase B: R logits, 256 tiles ----
    for (int tile = wv; tile < 256; tile += 8) {
      const int colg = tile * 16 + row16;
      Frag bfr;
      const unsigned short* wp = wbf + WS_WR + colg * 32 + bkb;
      bfr.q[0] = *(const uint4*)wp;
      bfr.q[1] = *(const uint4*)(wp + 16);

      f32x8 acc = {};
      acc = __builtin_amdgcn_wmma_f32_16x16x32_bf16(false, afr.v, false, bfr.v,
                                                    (short)0, acc, false, false);
      const float bv = br[colg];
      for (int r = 0; r < 8; ++r)
        Rbuf[(r + moff) * 4096 + colg] = acc[r] + bv;
    }
    __syncthreads();

    // ---- R softmax in place over each (s,h) row of 64, bank-skewed ----
    for (int rr = tid; rr < 1024; rr += 256) {
      int ss = rr >> 6, hh = rr & 63;
      float* base = Rbuf + ss * 4096 + hh * 64;
      int k0 = rr & 63;
      float mx = -1e30f;
      for (int i = 0; i < 64; ++i) { int k = (k0 + i) & 63; mx = fmaxf(mx, base[k]); }
      float den = 0.f;
      for (int i = 0; i < 64; ++i) { int k = (k0 + i) & 63; float e = __expf(base[k] - mx); base[k] = e; den += e; }
      float rd = fast_rcp(den);
      for (int i = 0; i < 64; ++i) { int k = (k0 + i) & 63; base[k] *= rd; }
    }
    __syncthreads();

    // ---- sequential recurrence: single wave, in-order LDS semantics ----
    if (wv == 0) {
      const int k0 = lane, k1 = lane + 32;
      for (int ss = 0; ss < 16; ++ss) {
        const float* rb = Rbuf + ss * 4096;
        float mn0 = MinS[ss * 64 + k0], mn1 = MinS[ss * 64 + k1];
        for (int hh = 0; hh < 64; ++hh) {
          float ch = Cst[hh];
          mn0 = fmaf(ch, rb[hh * 64 + k0], mn0);
          mn1 = fmaf(ch, rb[hh * 64 + k1], mn1);
        }
        float o0 = OgS[ss * 64 + k0], o1 = OgS[ss * 64 + k1];
        float cn0 = (1.f - o0) * mn0, cn1 = (1.f - o1) * mn1;
        size_t t = (size_t)(t0 + ss);
        out[64 + ((size_t)b * T_ + t) * H_ + k0] = cn0;
        out[64 + ((size_t)b * T_ + t) * H_ + k1] = cn1;
        Cst[k0] = cn0;  Cst[k1] = cn1;
        Mout[k0] = o0 * mn0;  Mout[k1] = o1 * mn1;
        __builtin_amdgcn_wave_barrier();
      }
    }
    __syncthreads();
  }

  // ---- fc head on final m_out ----
  if (tid < 64) Smax[tid] = Mout[tid] * Wfc[tid];
  __syncthreads();
  if (tid == 0) {
    float s = 0.f;
    for (int i = 0; i < 64; ++i) s += Smax[i];
    out[b] = s + bfc[0];
  }
}

extern "C" void kernel_launch(void* const* d_in, const int* in_sizes, int n_in,
                              void* d_out, int out_size, void* d_ws, size_t ws_size,
                              hipStream_t stream) {
  const float* x_m = (const float*)d_in[0];
  const float* x_a = (const float*)d_in[1];
  const float* Wj  = (const float*)d_in[2];
  const float* bj  = (const float*)d_in[3];
  const float* Wr  = (const float*)d_in[4];
  const float* br  = (const float*)d_in[5];
  const float* Wo  = (const float*)d_in[6];
  const float* bo  = (const float*)d_in[7];
  const float* Wfc = (const float*)d_in[8];
  const float* bfc = (const float*)d_in[9];
  float* out = (float*)d_out;
  unsigned short* wbf = (unsigned short*)d_ws;

  wconv_kernel<<<(W_TOTAL + 255) / 256, 256, 0, stream>>>(Wr, Wj, Wo, wbf);

  hipFuncSetAttribute(reinterpret_cast<const void*>(mclstm_kernel),
                      hipFuncAttributeMaxDynamicSharedMemorySize, (int)LDS_BYTES);
  mclstm_kernel<<<B_, 256, LDS_BYTES, stream>>>(x_m, x_a, bj, br, bo, Wfc, bfc, wbf, out);
}
